// CausalSelfAttention_21045339750475
// MI455X (gfx1250) — matrix operand order repaired and checked by reading, not compiled, not run on hardware
//
#include <hip/hip_runtime.h>

// ---------------------------------------------------------------------------
// CDNA5 (gfx1250) causal self-attention, bf16 WMMA + async-LDS path.
// B=4, T=2048, C=1024, H=16, HD=64.  wave32.
// ---------------------------------------------------------------------------

typedef __attribute__((ext_vector_type(16))) __bf16 v16bf;
typedef __attribute__((ext_vector_type(8)))  __bf16 v8bf;
typedef __attribute__((ext_vector_type(4)))  __bf16 v4bf;
typedef __attribute__((ext_vector_type(8)))  float  v8f;

#define TB   4
#define TT   2048
#define TC   1024
#define TH   16
#define THD  64
#define TM   (TB * TT)          // 8192 rows
#define TK   TC                 // 1024
#define TN   TC                 // 1024

static __device__ __forceinline__ v8f v8f_zero() {
    v8f z = {};
    return z;
}

// Async global->LDS B128 (ASYNCcnt path, §15.18.3 op 98, GVS addressing).
static __device__ __forceinline__ void async_ld_b128(unsigned lds_off,
                                                     unsigned gbyte_off,
                                                     const void* base) {
    asm volatile("global_load_async_to_lds_b128 %0, %1, %2"
                 :: "v"(lds_off), "v"(gbyte_off), "s"(base)
                 : "memory");
}
static __device__ __forceinline__ void wait_async_le1() {
    asm volatile("s_wait_asynccnt 0x1" ::: "memory");
}
static __device__ __forceinline__ void wait_async_le0() {
    asm volatile("s_wait_asynccnt 0x0" ::: "memory");
}

// 16x32 bf16 A-fragment (ISA 7.12.2): lane half selects K-subrange; two
// contiguous 8-element runs per lane.
static __device__ __forceinline__ v16bf load_afrag(const __bf16* run0,
                                                   const __bf16* run1) {
    v8bf lo = *reinterpret_cast<const v8bf*>(run0);
    v8bf hi = *reinterpret_cast<const v8bf*>(run1);
    v16bf a;
#pragma unroll
    for (int e = 0; e < 8; ++e) { a[e] = lo[e]; a[8 + e] = hi[e]; }
    return a;
}

// ---------------------------------------------------------------------------
// Kernel 0: one-shot fp32 -> bf16 conversion (x and the four weights).
// ---------------------------------------------------------------------------
__global__ __launch_bounds__(256) void cvt_f32_bf16_kernel(
    const float* __restrict__ src, __bf16* __restrict__ dst, int n4)
{
    const int i = blockIdx.x * blockDim.x + threadIdx.x;
    if (i < n4) {
        const float4 f = reinterpret_cast<const float4*>(src)[i];
        v4bf o;
        o[0] = (__bf16)f.x; o[1] = (__bf16)f.y;
        o[2] = (__bf16)f.z; o[3] = (__bf16)f.w;
        reinterpret_cast<v4bf*>(dst)[i] = o;
    }
}

// ---------------------------------------------------------------------------
// Kernel 1: QKV projection (all-bf16 operands).
// grid = (TM/32, TN/256, 3), block = 256 (8 waves, 2M x 4N wave grid).
// A-tile staged via async global->LDS B128, double-buffered.
// z=0 -> Q [bh,t,d];  z=1 -> K [bh,t,d];  z=2 -> V^T [bh,d,t] (packed stores).
// ---------------------------------------------------------------------------
__global__ __launch_bounds__(256) void qkv_proj_kernel(
    const __bf16* __restrict__ xb,
    const __bf16* __restrict__ Wqb, const float* __restrict__ bq,
    const __bf16* __restrict__ Wkb, const float* __restrict__ bk,
    const __bf16* __restrict__ Wvb, const float* __restrict__ bv,
    __bf16* __restrict__ qb, __bf16* __restrict__ kb, __bf16* __restrict__ vtb)
{
    const int z = blockIdx.z;
    const __bf16* W    = (z == 0) ? Wqb : (z == 1) ? Wkb : Wvb;
    const float*  bias = (z == 0) ? bq  : (z == 1) ? bk  : bv;

    const int tid   = threadIdx.x;
    const int lane  = tid & 31;
    const int wave  = tid >> 5;
    const int waveM = wave >> 2;
    const int waveN = wave & 3;
    const int mtile = blockIdx.x * 32;
    const int ntile = blockIdx.y * 256;
    const int lhalf = lane >> 4;
    const int l16   = lane & 15;
    const int khalf = lhalf * 8;

    __shared__ __bf16 lds_a[2][32][40];      // double-buffered 32x32 bf16 tile

    v8f acc[4];
#pragma unroll
    for (int i = 0; i < 4; ++i) acc[i] = v8f_zero();

    // Async staging: waves 0-3 (tid<128) each move one B128 per stage.
    const int srow = tid >> 2;               // 0..31
    const int sseg = (tid & 3) * 8;          // element offset 0/8/16/24
    const bool stager = (tid < 128);

    if (stager) {
        const unsigned goff = (unsigned)(((mtile + srow) * TK + 0 + sseg) * 2);
        async_ld_b128((unsigned)(uintptr_t)&lds_a[0][srow][sseg], goff, xb);
    }

    for (int k0 = 0, it = 0; k0 < TK; k0 += 32, ++it) {
        const int buf = it & 1;
        const bool more = (k0 + 32 < TK);
        if (stager && more) {
            const unsigned goff =
                (unsigned)(((mtile + srow) * TK + k0 + 32 + sseg) * 2);
            async_ld_b128((unsigned)(uintptr_t)&lds_a[buf ^ 1][srow][sseg],
                          goff, xb);
        }
        if (more) wait_async_le1(); else wait_async_le0();
        __syncthreads();                     // stage `it` visible to all waves

        const __bf16* arow = &lds_a[buf][waveM * 16 + l16][0];
        const v16bf afrag = load_afrag(arow + lhalf * 16, arow + lhalf * 16 + 8);

#pragma unroll
        for (int i = 0; i < 4; ++i) {
            const int n = ntile + waveN * 64 + i * 16 + l16;
            const __bf16* wrow = W + (size_t)n * TK + k0 + lhalf * 16;
            __builtin_prefetch(wrow + 32, 0, 1);       // global_prefetch_b8
            const v16bf bfrag = *reinterpret_cast<const v16bf*>(wrow);
            acc[i] = __builtin_amdgcn_wmma_f32_16x16x32_bf16(
                false, afrag, false, bfrag, (short)0, acc[i], false, false);
        }
        __syncthreads();                     // all reads of `buf` done
    }

    // Epilogue.
#pragma unroll
    for (int i = 0; i < 4; ++i) {
        const int n  = ntile + waveN * 64 + i * 16 + l16;
        const int h  = n >> 6;
        const int d  = n & 63;
        const float bn = bias[n];
        const int m0 = mtile + waveM * 16 + lhalf * 8;
        const int b  = m0 >> 11;
        const int t0 = m0 & 2047;
        const int bh = b * TH + h;
        if (z == 2) {
            v8bf pk;                         // 8 consecutive t for fixed (bh,d)
#pragma unroll
            for (int r = 0; r < 8; ++r) pk[r] = (__bf16)(acc[i][r] + bn);
            *reinterpret_cast<v8bf*>(
                vtb + ((size_t)bh * THD + d) * TT + t0) = pk;
        } else {
            __bf16* dst = (z == 0) ? qb : kb;
#pragma unroll
            for (int r = 0; r < 8; ++r)
                dst[((size_t)bh * TT + t0 + r) * THD + d] =
                    (__bf16)(acc[i][r] + bn);
        }
    }
}

// ---------------------------------------------------------------------------
// Kernel 2: causal flash attention.
// grid = (T/64, B*H), block = 128 (4 waves, one 16-query tile per wave).
// Per 32-key chunk: 4 WMMAs S=QK^T, online softmax, P via LDS, 4 WMMAs O+=PV.
// ---------------------------------------------------------------------------
__global__ __launch_bounds__(128) void attn_kernel(
    const __bf16* __restrict__ qb, const __bf16* __restrict__ kb,
    const __bf16* __restrict__ vtb, __bf16* __restrict__ yb)
{
    const int tid   = threadIdx.x;
    const int lane  = tid & 31;
    const int wave  = tid >> 5;
    const int bh    = blockIdx.y;
    const int q0    = blockIdx.x * 64 + wave * 16;
    const int lhalf = lane >> 4;
    const int l16   = lane & 15;
    const int khalf = lhalf * 8;
    const size_t rowbase = (size_t)bh * TT;

    __shared__ __bf16 pbuf[4][16][40];

    const __bf16* qrow = qb + (rowbase + q0 + l16) * THD;
    const v16bf aq0 = load_afrag(qrow + lhalf * 16,      qrow + lhalf * 16 + 8);
    const v16bf aq1 = load_afrag(qrow + 32 + lhalf * 16, qrow + 40 + lhalf * 16);

    v8f acc[4];
#pragma unroll
    for (int i = 0; i < 4; ++i) acc[i] = v8f_zero();
    float mrun[8], lrun[8];
#pragma unroll
    for (int r = 0; r < 8; ++r) { mrun[r] = -1e30f; lrun[r] = 0.0f; }

    const float scale = 0.125f;
    const int   kend  = q0 + 16;

    for (int kc = 0; kc < kend; kc += 32) {
        v8f s0 = v8f_zero(), s1 = v8f_zero();
        {
            const __bf16* kp = kb + (rowbase + kc + l16) * THD + lhalf * 16;
            const v16bf bk0 = *reinterpret_cast<const v16bf*>(kp);
            const v16bf bk1 = *reinterpret_cast<const v16bf*>(kp + 32);
            s0 = __builtin_amdgcn_wmma_f32_16x16x32_bf16(false, aq0, false, bk0,
                                                         (short)0, s0, false, false);
            s0 = __builtin_amdgcn_wmma_f32_16x16x32_bf16(false, aq1, false, bk1,
                                                         (short)0, s0, false, false);
        }
        {
            const __bf16* kp = kb + (rowbase + kc + 16 + l16) * THD + lhalf * 16;
            const v16bf bk0 = *reinterpret_cast<const v16bf*>(kp);
            const v16bf bk1 = *reinterpret_cast<const v16bf*>(kp + 32);
            s1 = __builtin_amdgcn_wmma_f32_16x16x32_bf16(false, aq0, false, bk0,
                                                         (short)0, s1, false, false);
            s1 = __builtin_amdgcn_wmma_f32_16x16x32_bf16(false, aq1, false, bk1,
                                                         (short)0, s1, false, false);
        }

        float alpha[8];
#pragma unroll
        for (int r = 0; r < 8; ++r) {
            const int qi = q0 + lhalf * 8 + r;
            float v0 = s0[r] * scale; if (kc + l16 > qi)      v0 = -1e30f;
            float v1 = s1[r] * scale; if (kc + 16 + l16 > qi) v1 = -1e30f;
            float t = fmaxf(v0, v1);
            t = fmaxf(t, __shfl_xor(t, 1, 32));
            t = fmaxf(t, __shfl_xor(t, 2, 32));
            t = fmaxf(t, __shfl_xor(t, 4, 32));
            t = fmaxf(t, __shfl_xor(t, 8, 32));
            const float mn = fmaxf(mrun[r], t);
            const float a  = __expf(mrun[r] - mn);
            const float p0 = __expf(v0 - mn);
            const float p1 = __expf(v1 - mn);
            float su = p0 + p1;
            su += __shfl_xor(su, 1, 32);
            su += __shfl_xor(su, 2, 32);
            su += __shfl_xor(su, 4, 32);
            su += __shfl_xor(su, 8, 32);
            lrun[r] = lrun[r] * a + su;
            mrun[r] = mn;
            alpha[r] = a;
            const int mr = lhalf * 8 + r;
            pbuf[wave][mr][l16]      = (__bf16)p0;
            pbuf[wave][mr][16 + l16] = (__bf16)p1;
        }
#pragma unroll
        for (int i = 0; i < 4; ++i)
#pragma unroll
            for (int r = 0; r < 8; ++r) acc[i][r] *= alpha[r];

        const __bf16* prow = &pbuf[wave][l16][0];
        const v16bf ap = load_afrag(prow + lhalf * 16, prow + lhalf * 16 + 8);
#pragma unroll
        for (int i = 0; i < 4; ++i) {
            const __bf16* vp =
                vtb + ((size_t)bh * THD + i * 16 + l16) * TT + kc + lhalf * 16;
            const v16bf bv16 = *reinterpret_cast<const v16bf*>(vp);
            acc[i] = __builtin_amdgcn_wmma_f32_16x16x32_bf16(
                false, ap, false, bv16, (short)0, acc[i], false, false);
        }
    }

    const int b = bh >> 4, h = bh & 15;
#pragma unroll
    for (int i = 0; i < 4; ++i) {
        const int c = h * THD + i * 16 + l16;
#pragma unroll
        for (int r = 0; r < 8; ++r) {
            const int t = q0 + lhalf * 8 + r;
            yb[((size_t)b * TT + t) * TC + c] = (__bf16)(acc[i][r] / lrun[r]);
        }
    }
}

// ---------------------------------------------------------------------------
// Kernel 3: output projection (bf16 A from ws, bf16 W, fp32 out + bias).
// ---------------------------------------------------------------------------
__global__ __launch_bounds__(256) void out_proj_kernel(
    const __bf16* __restrict__ yb, const __bf16* __restrict__ Wpb,
    const float* __restrict__ bp, float* __restrict__ out)
{
    const int tid   = threadIdx.x;
    const int lane  = tid & 31;
    const int wave  = tid >> 5;
    const int waveM = wave >> 2;
    const int waveN = wave & 3;
    const int mtile = blockIdx.x * 32;
    const int ntile = blockIdx.y * 256;
    const int lhalf = lane >> 4;
    const int l16   = lane & 15;
    const int khalf = lhalf * 8;

    __shared__ __bf16 lds_a[2][32][40];

    v8f acc[4];
#pragma unroll
    for (int i = 0; i < 4; ++i) acc[i] = v8f_zero();

    const int srow = tid >> 2;
    const int sseg = (tid & 3) * 8;
    const bool stager = (tid < 128);

    if (stager) {
        const unsigned goff = (unsigned)(((mtile + srow) * TK + 0 + sseg) * 2);
        async_ld_b128((unsigned)(uintptr_t)&lds_a[0][srow][sseg], goff, yb);
    }

    for (int k0 = 0, it = 0; k0 < TK; k0 += 32, ++it) {
        const int buf = it & 1;
        const bool more = (k0 + 32 < TK);
        if (stager && more) {
            const unsigned goff =
                (unsigned)(((mtile + srow) * TK + k0 + 32 + sseg) * 2);
            async_ld_b128((unsigned)(uintptr_t)&lds_a[buf ^ 1][srow][sseg],
                          goff, yb);
        }
        if (more) wait_async_le1(); else wait_async_le0();
        __syncthreads();

        const __bf16* arow = &lds_a[buf][waveM * 16 + l16][0];
        const v16bf afrag = load_afrag(arow + lhalf * 16, arow + lhalf * 16 + 8);

#pragma unroll
        for (int i = 0; i < 4; ++i) {
            const int n = ntile + waveN * 64 + i * 16 + l16;
            const __bf16* wrow = Wpb + (size_t)n * TK + k0 + lhalf * 16;
            __builtin_prefetch(wrow + 32, 0, 1);
            const v16bf bfrag = *reinterpret_cast<const v16bf*>(wrow);
            acc[i] = __builtin_amdgcn_wmma_f32_16x16x32_bf16(
                false, afrag, false, bfrag, (short)0, acc[i], false, false);
        }
        __syncthreads();
    }

#pragma unroll
    for (int i = 0; i < 4; ++i) {
        const int n  = ntile + waveN * 64 + i * 16 + l16;
        const float bn = bp[n];
#pragma unroll
        for (int r = 0; r < 8; ++r) {
            const int m = mtile + waveM * 16 + lhalf * 8 + r;
            out[(size_t)m * TN + n] = acc[i][r] + bn;
        }
    }
}

// ---------------------------------------------------------------------------
// Host launcher.
// d_in order: x, Wq, bq, Wk, bk, Wv, bv, Wp, bp
// ws (bf16 elems): xb | Q | K | V^T | Y | Wq | Wk | Wv | Wp   (88 MB)
// ---------------------------------------------------------------------------
extern "C" void kernel_launch(void* const* d_in, const int* in_sizes, int n_in,
                              void* d_out, int out_size, void* d_ws, size_t ws_size,
                              hipStream_t stream) {
    (void)in_sizes; (void)n_in; (void)out_size; (void)ws_size;

    const float* x  = (const float*)d_in[0];
    const float* Wq = (const float*)d_in[1];
    const float* bq = (const float*)d_in[2];
    const float* Wk = (const float*)d_in[3];
    const float* bk = (const float*)d_in[4];
    const float* Wv = (const float*)d_in[5];
    const float* bv = (const float*)d_in[6];
    const float* Wp = (const float*)d_in[7];
    const float* bp = (const float*)d_in[8];
    float* out = (float*)d_out;

    const size_t NBIG = (size_t)TM * TK;       // 8,388,608 elems
    const size_t NW   = (size_t)TN * TK;       // 1,048,576 elems
    __bf16* base = (__bf16*)d_ws;
    __bf16* xb  = base;
    __bf16* qb  = base + NBIG;
    __bf16* kb  = base + 2 * NBIG;
    __bf16* vtb = base + 3 * NBIG;
    __bf16* yb  = base + 4 * NBIG;
    __bf16* Wqb = base + 5 * NBIG;
    __bf16* Wkb = Wqb + NW;
    __bf16* Wvb = Wkb + NW;
    __bf16* Wpb = Wvb + NW;

    // One-shot fp32 -> bf16 conversions.
    {
        const int n4x = (int)(NBIG / 4), n4w = (int)(NW / 4);
        cvt_f32_bf16_kernel<<<(n4x + 255) / 256, 256, 0, stream>>>(x,  xb,  n4x);
        cvt_f32_bf16_kernel<<<(n4w + 255) / 256, 256, 0, stream>>>(Wq, Wqb, n4w);
        cvt_f32_bf16_kernel<<<(n4w + 255) / 256, 256, 0, stream>>>(Wk, Wkb, n4w);
        cvt_f32_bf16_kernel<<<(n4w + 255) / 256, 256, 0, stream>>>(Wv, Wvb, n4w);
        cvt_f32_bf16_kernel<<<(n4w + 255) / 256, 256, 0, stream>>>(Wp, Wpb, n4w);
    }

    qkv_proj_kernel<<<dim3(TM / 32, TN / 256, 3), 256, 0, stream>>>(
        xb, Wqb, bq, Wkb, bk, Wvb, bv, qb, kb, vtb);

    attn_kernel<<<dim3(TT / 64, TB * TH), 128, 0, stream>>>(qb, kb, vtb, yb);

    out_proj_kernel<<<dim3(TM / 32, TN / 256), 256, 0, stream>>>(yb, Wpb, bp, out);
}
